// ImprovedGate_89687507075959
// MI455X (gfx1250) — matrix operand order repaired and checked by reading, not compiled, not run on hardware
//
#include <hip/hip_runtime.h>
#include <math.h>

#define DD 2048
#define HH 256
#define H2 128
#define EE 64

typedef __attribute__((ext_vector_type(16))) __bf16 v16bf;
typedef __attribute__((ext_vector_type(8)))  float  v8f;
typedef __attribute__((ext_vector_type(4)))  unsigned u32x4;
typedef __attribute__((ext_vector_type(8)))  int      i32x8;
typedef __attribute__((ext_vector_type(4)))  int      i32x4;

union Frag { v16bf v; unsigned u[8]; uint4 q[2]; };

__device__ __forceinline__ unsigned packbf(float a, float b) {
  unsigned ua = __float_as_uint(a), ub = __float_as_uint(b);
  unsigned ha = (ua + 0x7FFFu + ((ua >> 16) & 1u)) >> 16;
  unsigned hb = ((ub + 0x7FFFu + ((ub >> 16) & 1u)) >> 16) << 16;
  return ha | hb;
}

__device__ __forceinline__ float gelu_exact(float x) {
  return 0.5f * x * (1.0f + erff(x * 0.70710678118654752440f));
}

// Issue one TDM copy of `nelem` 4-byte elements, global -> LDS (1-D tile).
// D# layout per CDNA5 ISA ch.8 (group0: count/lds_addr/global_addr/type=2,
// group1: data_size=4B, tensor_dim0=tile_dim0=nelem, stride=nelem).
// 6-arg builtin form (clang-23 / therock-10.0 signature).
__device__ __forceinline__ void tdm_load(const void* gsrc, void* ldst, unsigned nelem) {
  unsigned long long ga = (unsigned long long)gsrc;
  unsigned la = (unsigned)(unsigned long long)ldst;   // flat LDS addr low 32 = LDS offset
  u32x4 g0;
  g0[0] = 1u;                                          // count=1 (user D#)
  g0[1] = la;                                          // lds_addr
  g0[2] = (unsigned)(ga & 0xFFFFFFFFu);                // global_addr[31:0]
  g0[3] = (unsigned)((ga >> 32) & 0x1FFFFFFu) | (2u << 30); // global_addr[56:32] | type=2
  i32x8 g1;
  g1[0] = (int)(2u << 16);                             // workgroup_mask=0, data_size=2 (4B)
  g1[1] = (int)((nelem & 0xFFFFu) << 16);              // tensor_dim0[15:0]
  g1[2] = (int)(((nelem >> 16) & 0xFFFFu) | (1u << 16)); // tensor_dim0[31:16] | tensor_dim1=1
  g1[3] = (int)((nelem & 0xFFFFu) << 16);              // tensor_dim1[31:16]=0 | tile_dim0
  g1[4] = 0;                                           // tile_dim1=0, tile_dim2=0
  g1[5] = (int)nelem;                                  // tensor_dim0_stride[31:0]
  g1[6] = 0;
  g1[7] = 0;
  i32x4 z4 = {0, 0, 0, 0};
  i32x8 z8 = {0, 0, 0, 0, 0, 0, 0, 0};
  __builtin_amdgcn_tensor_load_to_lds(g0, g1, z4, z4, z8, 0);
}

// ---- prep: pack w1/w2/w3 to bf16 pairs in exact WMMA-B fragment order ----
// ws layout (dwords): w2f[16384] | w3f[4096] | w1f[262144]
// block layout per (ks,t): 256 dwords; lane L owns granules (2L)^a,(2L+1)^a (a=(L>>3)&3)
__global__ void pack_weights_kernel(const float* __restrict__ w1,
                                    const float* __restrict__ w2,
                                    const float* __restrict__ w3,
                                    unsigned* __restrict__ ws) {
  unsigned* w2f = ws;
  unsigned* w3f = ws + 16384;
  unsigned* w1f = ws + 20480;
  int idx = blockIdx.x * 256 + threadIdx.x;
  if (idx < 262144) {                                  // w1: 64 ks x 16 t
    int ks = idx >> 12, r = idx & 4095, t = r >> 8, r2 = r & 255;
    int lane = r2 >> 3, p = r2 & 7;
    int n = t * 16 + (lane & 15), k = ks * 32 + ((p >= 4) ? 16 : 0) + (lane >> 4) * 8 + 2 * (p & 3);
    unsigned v = packbf(w1[k * HH + n], w1[(k + 1) * HH + n]);
    int a = (lane >> 3) & 3, dw = (((2 * lane + (p >> 2)) ^ a) << 2) | (p & 3);
    w1f[ks * 4096 + t * 256 + dw] = v;
  } else if (idx < 262144 + 16384) {                   // w2: 8 ks x 8 t
    int j = idx - 262144;
    int ks = j >> 11, r = j & 2047, t = r >> 8, r2 = r & 255;
    int lane = r2 >> 3, p = r2 & 7;
    int n = t * 16 + (lane & 15), k = ks * 32 + ((p >= 4) ? 16 : 0) + (lane >> 4) * 8 + 2 * (p & 3);
    unsigned v = packbf(w2[k * H2 + n], w2[(k + 1) * H2 + n]);
    int a = (lane >> 3) & 3, dw = (((2 * lane + (p >> 2)) ^ a) << 2) | (p & 3);
    w2f[ks * 2048 + t * 256 + dw] = v;
  } else if (idx < 262144 + 16384 + 4096) {            // w3: 4 ks x 4 t
    int j = idx - 262144 - 16384;
    int ks = j >> 10, r = j & 1023, t = r >> 8, r2 = r & 255;
    int lane = r2 >> 3, p = r2 & 7;
    int n = t * 16 + (lane & 15), k = ks * 32 + ((p >= 4) ? 16 : 0) + (lane >> 4) * 8 + 2 * (p & 3);
    unsigned v = packbf(w3[k * EE + n], w3[(k + 1) * EE + n]);
    int a = (lane >> 3) & 3, dw = (((2 * lane + (p >> 2)) ^ a) << 2) | (p & 3);
    w3f[ks * 1024 + t * 256 + dw] = v;
  }
}

__global__ __launch_bounds__(128, 1) void router_fused_kernel(
    const float* __restrict__ x,  const float* __restrict__ b1,
    const float* __restrict__ g1, const float* __restrict__ be1,
    const float* __restrict__ b2, const float* __restrict__ g2,
    const float* __restrict__ be2, const float* __restrict__ b3,
    const float* __restrict__ temp, const unsigned* __restrict__ ws,
    float* __restrict__ out_gates, int* __restrict__ out_idx, float* __restrict__ out_logits)
{
  extern __shared__ char smem[];
  unsigned* w2L  = (unsigned*)(smem);            // 64KB fragment-ordered
  unsigned* w3L  = (unsigned*)(smem + 65536);    // 16KB
  unsigned* w1L0 = (unsigned*)(smem + 81920);    // 16KB (double buffer)
  unsigned* w1L1 = (unsigned*)(smem + 98304);    // 16KB
  const int tid  = threadIdx.x;
  const int lane = tid & 31;
  const int wv   = tid >> 5;
  const int m    = lane & 15;
  const int hi   = lane >> 4;
  const int axor = (lane >> 3) & 3;
  char* wscr = smem + 114688 + wv * 25088;       // padded per-wave scratch
  float*    h1f = (float*)wscr;                  // [16][260] f32
  unsigned* h1b = (unsigned*)(wscr + 16640);     // [16][132] bf16x2
  float*    h2f = (float*)wscr;                  // reuse: [16][132] f32
  unsigned* h2b = (unsigned*)(wscr + 8448);      // [16][68]
  float*    lgs = (float*)(wscr + 12800);        // [16][68]

  const unsigned* w1f = ws + 20480;
  const long blockbase = (long)blockIdx.x * 64;
  const long rowg = blockbase + wv * 16 + m;

  // ---- TDM: weights (w2+w3 = 80KB) and w1 chunk 0 into LDS ----
  if (wv == 0) {
    tdm_load(ws, w2L, 20480);
    tdm_load(w1f, w1L0, 4096);
    __builtin_amdgcn_s_wait_tensorcnt(0);
  }
  __syncthreads();

  // ================= GEMM1: x[16 x 2048] @ w1[2048 x 256] =================
  v8f acc1[16];
  #pragma unroll
  for (int t = 0; t < 16; t++) {
    float bv = b1[t * 16 + m];
    acc1[t] = (v8f){bv, bv, bv, bv, bv, bv, bv, bv};
  }
  const float* xrow = x + rowg * DD;
  for (int ks = 0; ks < 64; ++ks) {
    if (wv == 0 && ks + 1 < 64)
      tdm_load(w1f + (ks + 1) * 4096, (ks & 1) ? w1L0 : w1L1, 4096);
    const unsigned* cur = (ks & 1) ? w1L1 : w1L0;
    const int k0 = ks * 32;
    __builtin_prefetch(xrow + k0 + 64, 0, 0);
    Frag a;
    #pragma unroll
    for (int p = 0; p < 8; p++) {                // documented 16-bit A layout
      int k = k0 + ((p >= 4) ? 16 : 0) + (p & 3) * 2 + hi * 8;
      float2 f = *(const float2*)(xrow + k);
      a.u[p] = packbf(f.x, f.y);
    }
    #pragma unroll
    for (int t = 0; t < 16; t++) {
      const uint4* bb = (const uint4*)(cur + t * 256);
      Frag bf;
      bf.q[0] = bb[(2 * lane) ^ axor];
      bf.q[1] = bb[(2 * lane + 1) ^ axor];
      acc1[t] = __builtin_amdgcn_wmma_f32_16x16x32_bf16(
          false, a.v, false, bf.v, (short)0, acc1[t], false, false);
    }
    if (wv == 0) __builtin_amdgcn_s_wait_tensorcnt(0);
    __syncthreads();
  }

  // ---- spill C (N across lanes, M = r + 8*hi) ----
  #pragma unroll
  for (int t = 0; t < 16; t++)
    #pragma unroll
    for (int r = 0; r < 8; r++)
      h1f[(r + 8 * hi) * 260 + t * 16 + m] = acc1[t][r];
  asm volatile("" ::: "memory");

  // ---- LN1 + exact GELU -> packed bf16 (2 lanes per row) ----
  {
    const int cb = hi * 128;
    float s = 0.f, s2 = 0.f;
    for (int c = 0; c < 128; c++) { float v = h1f[m * 260 + cb + c]; s += v; s2 += v * v; }
    s  += __shfl_xor(s, 16, 32);
    s2 += __shfl_xor(s2, 16, 32);
    float mu = s * (1.0f / HH), var = s2 * (1.0f / HH) - mu * mu;
    float rstd = rsqrtf(var + 1e-5f);
    for (int c = 0; c < 128; c += 2) {
      int cc = cb + c;
      float v0 = (h1f[m * 260 + cc]     - mu) * rstd * g1[cc]     + be1[cc];
      float v1 = (h1f[m * 260 + cc + 1] - mu) * rstd * g1[cc + 1] + be1[cc + 1];
      h1b[m * 132 + (cc >> 1)] = packbf(gelu_exact(v0), gelu_exact(v1));
    }
  }
  asm volatile("" ::: "memory");

  // ================= GEMM2: h1[16x256] @ w2[256x128] =================
  v8f acc2[8];
  #pragma unroll
  for (int t = 0; t < 8; t++) {
    float bv = b2[t * 16 + m];
    acc2[t] = (v8f){bv, bv, bv, bv, bv, bv, bv, bv};
  }
  for (int ks = 0; ks < 8; ++ks) {
    Frag a;
    a.q[0] = *(const uint4*)(h1b + m * 132 + ks * 16 + hi * 4);
    a.q[1] = *(const uint4*)(h1b + m * 132 + ks * 16 + hi * 4 + 8);
    #pragma unroll
    for (int t = 0; t < 8; t++) {
      const uint4* bb = (const uint4*)(w2L + (ks * 8 + t) * 256);
      Frag bf;
      bf.q[0] = bb[(2 * lane) ^ axor];
      bf.q[1] = bb[(2 * lane + 1) ^ axor];
      acc2[t] = __builtin_amdgcn_wmma_f32_16x16x32_bf16(
          false, a.v, false, bf.v, (short)0, acc2[t], false, false);
    }
  }
  #pragma unroll
  for (int t = 0; t < 8; t++)
    #pragma unroll
    for (int r = 0; r < 8; r++)
      h2f[(r + 8 * hi) * 132 + t * 16 + m] = acc2[t][r];
  asm volatile("" ::: "memory");

  // ---- LN2 + GELU ----
  {
    const int cb = hi * 64;
    float s = 0.f, s2 = 0.f;
    for (int c = 0; c < 64; c++) { float v = h2f[m * 132 + cb + c]; s += v; s2 += v * v; }
    s  += __shfl_xor(s, 16, 32);
    s2 += __shfl_xor(s2, 16, 32);
    float mu = s * (1.0f / H2), var = s2 * (1.0f / H2) - mu * mu;
    float rstd = rsqrtf(var + 1e-5f);
    for (int c = 0; c < 64; c += 2) {
      int cc = cb + c;
      float v0 = (h2f[m * 132 + cc]     - mu) * rstd * g2[cc]     + be2[cc];
      float v1 = (h2f[m * 132 + cc + 1] - mu) * rstd * g2[cc + 1] + be2[cc + 1];
      h2b[m * 68 + (cc >> 1)] = packbf(gelu_exact(v0), gelu_exact(v1));
    }
  }
  asm volatile("" ::: "memory");

  // ================= GEMM3: h2[16x128] @ w3[128x64] =================
  v8f acc3[4];
  #pragma unroll
  for (int t = 0; t < 4; t++) {
    float bv = b3[t * 16 + m];
    acc3[t] = (v8f){bv, bv, bv, bv, bv, bv, bv, bv};
  }
  for (int ks = 0; ks < 4; ++ks) {
    Frag a;
    a.q[0] = *(const uint4*)(h2b + m * 68 + ks * 16 + hi * 4);
    a.q[1] = *(const uint4*)(h2b + m * 68 + ks * 16 + hi * 4 + 8);
    #pragma unroll
    for (int t = 0; t < 4; t++) {
      const uint4* bb = (const uint4*)(w3L + (ks * 4 + t) * 256);
      Frag bf;
      bf.q[0] = bb[(2 * lane) ^ axor];
      bf.q[1] = bb[(2 * lane + 1) ^ axor];
      acc3[t] = __builtin_amdgcn_wmma_f32_16x16x32_bf16(
          false, a.v, false, bf.v, (short)0, acc3[t], false, false);
    }
  }

  // ---- temperature-scaled logits -> LDS + global ----
  const float tscale = 1.0f / fmaxf(temp[0], 0.1f);
  #pragma unroll
  for (int t = 0; t < 4; t++)
    #pragma unroll
    for (int r = 0; r < 8; r++) {
      int col = t * 16 + m, ml = r + 8 * hi;
      float v = acc3[t][r] * tscale;
      lgs[ml * 68 + col] = v;
      out_logits[(blockbase + wv * 16 + ml) * EE + col] = v;
    }
  asm volatile("" ::: "memory");

  // ---- top-2 + softmax + renorm; 2 lanes per row ----
  {
    const int cb = hi * 32;
    const long rg2 = blockbase + wv * 16 + m;
    float v1 = -INFINITY, v2 = -INFINITY; int i1 = -1, i2 = -1;
    for (int c = 0; c < 32; c++) {
      float v = lgs[m * 68 + cb + c]; int idx = cb + c;
      if (v > v1)      { v2 = v1; i2 = i1; v1 = v; i1 = idx; }
      else if (v > v2) { v2 = v;  i2 = idx; }
    }
    float pv1 = __shfl_xor(v1, 16, 32), pv2 = __shfl_xor(v2, 16, 32);
    int   pi1 = __shfl_xor(i1, 16, 32), pi2 = __shfl_xor(i2, 16, 32);
    if (pv1 > v1 || (pv1 == v1 && pi1 < i1)) { v2 = v1; i2 = i1; v1 = pv1; i1 = pi1; }
    else if (pv1 > v2 || (pv1 == v2 && pi1 < i2)) { v2 = pv1; i2 = pi1; }
    if (pv2 > v1 || (pv2 == v1 && pi2 < i1)) { v2 = v1; i2 = i1; v1 = pv2; i1 = pi2; }
    else if (pv2 > v2 || (pv2 == v2 && pi2 < i2)) { v2 = pv2; i2 = pi2; }

    float e2 = expf(v2 - v1);
    float ssum = 1.0f + e2;
    float t0 = 1.0f / ssum, t1 = e2 / ssum;
    float dn = t0 + t1 + 1e-8f;
    float g0 = t0 / dn, gg1 = t1 / dn;

    float4 z = {0.f, 0.f, 0.f, 0.f};
    float* gout = out_gates + rg2 * EE + cb;
    for (int c = 0; c < 32; c += 4) *(float4*)(gout + c) = z;
    if (hi == 0) {
      out_idx[rg2 * 2]     = i1;
      out_idx[rg2 * 2 + 1] = i2;
      out_gates[rg2 * EE + i1] = g0;
      out_gates[rg2 * EE + i2] = gg1;
    }
  }
}

extern "C" void kernel_launch(void* const* d_in, const int* in_sizes, int n_in,
                              void* d_out, int out_size, void* d_ws, size_t ws_size,
                              hipStream_t stream) {
  const float* x   = (const float*)d_in[0];
  const float* w1  = (const float*)d_in[1];
  const float* b1  = (const float*)d_in[2];
  const float* g1  = (const float*)d_in[3];
  const float* be1 = (const float*)d_in[4];
  const float* w2  = (const float*)d_in[5];
  const float* b2  = (const float*)d_in[6];
  const float* g2  = (const float*)d_in[7];
  const float* be2 = (const float*)d_in[8];
  const float* w3  = (const float*)d_in[9];
  const float* b3  = (const float*)d_in[10];
  const float* tp  = (const float*)d_in[11];

  const long N = in_sizes[0] / DD;               // 16384
  float* gates  = (float*)d_out;
  int*   idx    = (int*)((float*)d_out + (size_t)N * EE);
  float* logits = (float*)d_out + (size_t)N * EE + (size_t)N * 2;
  unsigned* ws  = (unsigned*)d_ws;               // needs 80KB + 1MB

  pack_weights_kernel<<<dim3(1104), dim3(256), 0, stream>>>(w1, w2, w3, ws);

  const size_t shmem = 215040;                   // 210 KB dynamic LDS
  (void)hipFuncSetAttribute((const void*)router_fused_kernel,
                            hipFuncAttributeMaxDynamicSharedMemorySize, (int)shmem);
  dim3 grid((unsigned)(N / 64)), block(128);
  router_fused_kernel<<<grid, block, shmem, stream>>>(
      x, b1, g1, be1, b2, g2, be2, b3, tp, ws, gates, idx, logits);
}
